// LinearAttention_14027363189490
// MI455X (gfx1250) — compile-verified, matrix operand order checked by reading, not run on hardware
//
#include <hip/hip_runtime.h>
#include <hip/hip_bf16.h>

// ---------------------------------------------------------------------------
// LinearAttention for MI455X (gfx1250, wave32).
// bf16 WMMA (v_wmma_f32_16x16x32_bf16) for QKV GEMM, context GEMM (k v^T),
// and output GEMM; fp32 VALU for softmax / RMSNorm. Intermediates bf16 in ws.
// v-row staging in the attention kernel uses the CDNA5 async global->LDS
// path (ASYNCcnt) when the toolchain exposes it.
// ---------------------------------------------------------------------------

typedef __attribute__((ext_vector_type(8)))  float  v8f;
typedef __attribute__((ext_vector_type(16))) __bf16 v16bf;
typedef __attribute__((ext_vector_type(8)))  __bf16 v8bf;

#if defined(__gfx1250__) && __has_builtin(__builtin_amdgcn_global_load_async_to_lds_b128)
#define LA_ASYNC 1
typedef int la_v4i __attribute__((__vector_size__(16)));
typedef __attribute__((address_space(1))) la_v4i la_as1_v4i;   // global
typedef __attribute__((address_space(3))) la_v4i la_as3_v4i;   // LDS
#endif

#define NBATCH 32
#define CDIM   256
#define NPOS   4096
#define HEADS  4
#define DHEAD  32
#define HID    128
#define OQKV   384
#define NMEM   4

static __device__ __forceinline__ v16bf cat8(v8bf lo, v8bf hi) {
  return __builtin_shufflevector(lo, hi, 0,1,2,3,4,5,6,7,8,9,10,11,12,13,14,15);
}

// ---------------------------------------------------------------------------
// Kernel 0: repack W_qkv (384x256) and W_out (256x128) f32 row-major into
// bf16 WMMA B-fragment order. For tile (ob over N/16, kb over K/32):
// lane l holds column N = l&15, K-chunk base (l>=16 ? 8 : 0); 16 bf16 values
// in VGPR order: v<4 -> K = 2v,2v+1 ; v>=4 -> K = 16+2(v-4)+{0,1}.
// ---------------------------------------------------------------------------
__global__ void la_prep_weights(const float* __restrict__ Wq,
                                const float* __restrict__ Wo,
                                __bf16* __restrict__ dq,
                                __bf16* __restrict__ dw) {
  int slot = blockIdx.x * 256 + threadIdx.x;   // 8192 slots total
  const float* W;
  __bf16* dst;
  int lane, kb, ob, ccols;
  if (slot < 6144) {                            // W_qkv: 24 ob x 8 kb x 32 lanes
    lane = slot & 31; int tile = slot >> 5;
    kb = tile & 7; ob = tile >> 3;
    W = Wq; dst = dq + (size_t)slot * 16; ccols = 256;
  } else {                                      // W_out: 16 ob x 4 kb x 32 lanes
    int s2 = slot - 6144;
    lane = s2 & 31; int tile = s2 >> 5;
    kb = tile & 3; ob = tile >> 2;
    W = Wo; dst = dw + (size_t)s2 * 16; ccols = 128;
  }
  int n = lane & 15, hi = lane >> 4;
  const float* wrow = W + (size_t)(ob * 16 + n) * ccols + kb * 32 + hi * 8;
#pragma unroll
  for (int v = 0; v < 8; ++v) {
    int kbase = (v < 4) ? 2 * v : 16 + 2 * (v - 4);
    dst[v * 2 + 0] = (__bf16)wrow[kbase + 0];
    dst[v * 2 + 1] = (__bf16)wrow[kbase + 1];
  }
}

// ---------------------------------------------------------------------------
// Kernel 1: RMSNorm(x) + QKV projection.
// Grid (32 pos-tiles, 32 batches), 256 threads (8 waves).
// LDS: xn[128 pos][264 c] bf16 + reduce scratch = 69120 B (>64K opt-in; WGP
// has 320 KB). Each wave owns pos-block pb == wave: A-fragments loaded once
// into registers (v16bf afrag[8]) and reused across all 24 o-blocks.
// Output qkv bf16 laid out [b][o=384][n=4096].
// ---------------------------------------------------------------------------
#define XSTR 264
__global__ void la_qkv(const float* __restrict__ x,
                       const float* __restrict__ g1,
                       const __bf16* __restrict__ wq,
                       __bf16* __restrict__ qkv) {
  const int b = blockIdx.y, tile = blockIdx.x;
  const int t = threadIdx.x, lane = t & 31, wave = t >> 5;
  const int pbase = tile * 128;
  extern __shared__ char smem[];
  __bf16* xn   = (__bf16*)smem;                     // 128*264*2 = 67584
  float*  red  = (float*)(smem + 67584);            // 256 f32
  float*  rinv = (float*)(smem + 68608);            // 128 f32

  // warm L2/L0 with this block's weight fragments (global_prefetch_b8)
  __builtin_prefetch(wq + (size_t)t * 384, 0, 1);

  const int px = t & 127, half = t >> 7;
  const float* xb = x + (size_t)b * CDIM * NPOS + pbase + px;

  // pass 1: sum of squares over channels (each position split across 2 threads)
  float s = 0.f;
  for (int c = half * 128; c < half * 128 + 128; ++c) {
    float v = xb[(size_t)c * NPOS];
    s += v * v;
  }
  red[t] = s;
  __syncthreads();
  if (half == 0) {
    float nrm = sqrtf(red[px] + red[px + 128]);
    rinv[px] = 16.0f / fmaxf(nrm, 1e-12f);          // sqrt(DIM) / ||x||
  }
  __syncthreads();
  // pass 2: normalized bf16 tile into LDS, [pos][c]
  {
    float r = rinv[px];
    for (int c = half * 128; c < half * 128 + 128; ++c) {
      float v = xb[(size_t)c * NPOS] * r * g1[c];
      xn[px * XSTR + c] = (__bf16)v;
    }
  }
  __syncthreads();

  // GEMM: wave owns pb == wave. Load the 8 A-fragments once.
  const int cofs = (lane & 16) ? 8 : 0;
  const int arow = wave * 16 + (lane & 15);
  v16bf afrag[8];
#pragma unroll
  for (int kb = 0; kb < 8; ++kb) {
    const __bf16* ap = &xn[arow * XSTR + kb * 32 + cofs];
    afrag[kb] = cat8(*(const v8bf*)ap, *(const v8bf*)(ap + 16));
  }
  const int n_ = lane & 15, mbase = (lane >> 4) * 8;
  for (int ob = 0; ob < 24; ++ob) {
    v8f acc = {0.f, 0.f, 0.f, 0.f, 0.f, 0.f, 0.f, 0.f};
#pragma unroll
    for (int kb = 0; kb < 8; ++kb) {
      const __bf16* bp = wq + ((size_t)((ob * 8 + kb) * 32 + lane)) * 16;
      v16bf bm = cat8(*(const v8bf*)bp, *(const v8bf*)(bp + 8));
      acc = __builtin_amdgcn_wmma_f32_16x16x32_bf16(false, afrag[kb], false, bm,
                                                    (short)0, acc, false, false);
    }
    // D: VGPR r -> (M = r or 8+r by lane half, N = lane&15)
    __bf16* outp = qkv + ((size_t)b * OQKV + ob * 16 + n_) * NPOS
                       + pbase + wave * 16 + mbase;
#pragma unroll
    for (int r = 0; r < 8; ++r) outp[r] = (__bf16)acc[r];
  }
}

// ---------------------------------------------------------------------------
// Kernel 2: linear attention core per (head, batch). 256 threads (8 waves).
// Step 1: per-row online softmax stats via wave32 __shfl_xor reductions.
// Step 2: ctx = softmax(k) v^T as four 16x16 WMMA tiles; each tile split
//         across 2 waves by 128-column parity, partials combined in LDS.
//         v rows staged by the async global->LDS engine (ASYNCcnt) while
//         VALU computes the exp-normalized k rows.
// Step 3: q softmax over d + ctx apply per position (fp32 VALU).
// LDS 46336 B.
// ---------------------------------------------------------------------------
#define KSTR 264
__global__ void la_attn(const __bf16* __restrict__ qkv,
                        const float* __restrict__ memkv,
                        __bf16* __restrict__ attout) {
  const int h = blockIdx.x, b = blockIdx.y;
  const int t = threadIdx.x, lane = t & 31, wave = t >> 5;
  extern __shared__ char smem[];
  __bf16* ekb  = (__bf16*)smem;                     // 32 x 264 bf16 = 16896
  __bf16* vvb  = (__bf16*)(smem + 16896);           // 32 x 264 bf16 = 16896
  float*  rowm = (float*)(smem + 33792);            // 32
  float*  rows = rowm + 32;                         // 32
  float*  ctx  = rows + 32;                         // 32*32 f32
  float*  ctxp = ctx + 1024;                        // 2 x 1024 f32
                                                    // end = 46336

  const __bf16* qb = qkv + ((size_t)b * OQKV +       h * DHEAD) * NPOS;
  const __bf16* kb = qkv + ((size_t)b * OQKV + 128 + h * DHEAD) * NPOS;
  const __bf16* vb = qkv + ((size_t)b * OQKV + 256 + h * DHEAD) * NPOS;
  const float* mk = memkv + (size_t)(0 * HEADS + h) * DHEAD * NMEM;
  const float* mv = memkv + (size_t)(1 * HEADS + h) * DHEAD * NMEM;

  // ---- step 1: per-row (d) online max/sum over n = {mem, 0..4095} ----
  for (int r = 0; r < 4; ++r) {
    int d = wave * 4 + r;
    const __bf16* kr = kb + (size_t)d * NPOS;
    float m = -3.4e38f, sum = 0.f;
    for (int n = lane; n < NPOS; n += 32) {
      float v = (float)kr[n];
      float nm = fmaxf(m, v);
      sum = sum * __expf(m - nm) + __expf(v - nm);
      m = nm;
    }
#pragma unroll
    for (int off = 16; off > 0; off >>= 1) {
      float m2 = __shfl_xor(m, off);
      float s2 = __shfl_xor(sum, off);
      float nm = fmaxf(m, m2);
      sum = sum * __expf(m - nm) + s2 * __expf(m2 - nm);
      m = nm;
    }
    if (lane == 0) {
#pragma unroll
      for (int j = 0; j < NMEM; ++j) {
        float v = mk[d * NMEM + j];
        float nm = fmaxf(m, v);
        sum = sum * __expf(m - nm) + __expf(v - nm);
        m = nm;
      }
      rowm[d] = m; rows[d] = sum;
    }
  }
  __syncthreads();

  // ---- step 2: ctx[d][e] = sum_n softmax(k)[d,n] * v[e,n] via WMMA ----
  // 4 ctx tiles (db,eb in {0,1}); tile = wave&3, n-parity half = wave>>2.
  const int db = (wave >> 1) & 1, eb = wave & 1, hw = wave >> 2;
  const int cofs = (lane & 16) ? 8 : 0;
  const int arow = db * 16 + (lane & 15);
  const int brow = eb * 16 + (lane & 15);
  v8f acc = {0.f, 0.f, 0.f, 0.f, 0.f, 0.f, 0.f, 0.f};
  for (int c0 = 0; c0 < NPOS; c0 += 256) {
#ifdef LA_ASYNC
    // async DMA: each wave copies rows {wave, wave+8, wave+16, wave+24};
    // each lane moves 16 B (8 bf16) per op. Overlaps with the exp work below.
    {
      const __bf16* g = vb + (size_t)wave * NPOS + c0 + lane * 8;
      __bf16* l = vvb + wave * KSTR + lane * 8;
#pragma unroll
      for (int rr = 0; rr < 4; ++rr) {
        __builtin_amdgcn_global_load_async_to_lds_b128(
            (la_as1_v4i*)(g + (size_t)rr * 8 * NPOS),
            (la_as3_v4i*)(l + rr * 8 * KSTR), 0, 0);
      }
    }
    for (int i = t; i < 32 * 256; i += 256) {
      int r = i >> 8, n = i & 255;
      ekb[r * KSTR + n] =
          (__bf16)(__expf((float)kb[(size_t)r * NPOS + c0 + n] - rowm[r]) / rows[r]);
    }
#if __has_builtin(__builtin_amdgcn_s_wait_asynccnt)
    __builtin_amdgcn_s_wait_asynccnt(0);
#else
    asm volatile("s_wait_asynccnt 0x0" ::: "memory");
#endif
#else
    // stage 256-column superchunk: ek (exp-normalized, bf16) and v
    for (int i = t; i < 32 * 256; i += 256) {
      int r = i >> 8, n = i & 255;
      ekb[r * KSTR + n] =
          (__bf16)(__expf((float)kb[(size_t)r * NPOS + c0 + n] - rowm[r]) / rows[r]);
      vvb[r * KSTR + n] = vb[(size_t)r * NPOS + c0 + n];
    }
#endif
    __syncthreads();
#pragma unroll
    for (int kk = 0; kk < 4; ++kk) {
      const __bf16* ap = &ekb[arow * KSTR + hw * 128 + kk * 32 + cofs];
      v16bf a = cat8(*(const v8bf*)ap, *(const v8bf*)(ap + 16));
      const __bf16* bp = &vvb[brow * KSTR + hw * 128 + kk * 32 + cofs];
      v16bf bm = cat8(*(const v8bf*)bp, *(const v8bf*)(bp + 16));
      acc = __builtin_amdgcn_wmma_f32_16x16x32_bf16(false, a, false, bm,
                                                    (short)0, acc, false, false);
    }
    __syncthreads();
  }
  {
    int n_ = lane & 15, mbase = (lane >> 4) * 8;
    float* cp = ctxp + hw * 1024;
#pragma unroll
    for (int r = 0; r < 8; ++r)
      cp[(db * 16 + mbase + r) * 32 + eb * 16 + n_] = acc[r];
  }
  __syncthreads();
  // combine the two n-parity partials + mem_kv contribution
  for (int i = t; i < 1024; i += 256) {
    int d2 = i >> 5, e2 = i & 31;
    float a = ctxp[i] + ctxp[1024 + i];
    float rm = rowm[d2], rs = rows[d2];
#pragma unroll
    for (int j = 0; j < NMEM; ++j)
      a += (__expf(mk[d2 * NMEM + j] - rm) / rs) * mv[e2 * NMEM + j];
    ctx[i] = a;
  }
  __syncthreads();

  // ---- step 3: out[e,p] = sum_d ctx[d][e] * (softmax_d(q)*scale)[d,p] ----
  const float scale = 0.17677669529663687f;   // DHEAD^-0.5
  for (int it = 0; it < 16; ++it) {
    int p = it * 256 + t;
    float qv[32];
    float qm = -3.4e38f;
#pragma unroll
    for (int dd = 0; dd < 32; ++dd) {
      qv[dd] = (float)qb[(size_t)dd * NPOS + p];
      qm = fmaxf(qm, qv[dd]);
    }
    float qs = 0.f;
#pragma unroll
    for (int dd = 0; dd < 32; ++dd) { qv[dd] = __expf(qv[dd] - qm); qs += qv[dd]; }
    float inv = scale / qs;
    for (int e = 0; e < 32; ++e) {
      float a = 0.f;
#pragma unroll
      for (int dd = 0; dd < 32; ++dd) a += ctx[dd * 32 + e] * qv[dd];
      attout[((size_t)b * HID + h * DHEAD + e) * NPOS + p] = (__bf16)(a * inv);
    }
  }
}

// ---------------------------------------------------------------------------
// Kernel 3: output projection y = W_out * attout + b_out (bf16 WMMA).
// Grid (32 pos-tiles, 32 batches), 256 threads. LDS 34816 B.
// Wave owns pos-block pb == wave; A-fragments (afrag[4]) loaded once and
// reused across all 16 o-blocks.
// ---------------------------------------------------------------------------
#define ASTR 136
__global__ void la_proj(const __bf16* __restrict__ att,
                        const __bf16* __restrict__ wo,
                        const float* __restrict__ bout,
                        __bf16* __restrict__ y) {
  const int b = blockIdx.y, tile = blockIdx.x;
  const int t = threadIdx.x, lane = t & 31, wave = t >> 5;
  const int pbase = tile * 128;
  extern __shared__ char smem[];
  __bf16* as = (__bf16*)smem;                       // 128 pos x 136 ch

  __builtin_prefetch(wo + (size_t)t * 128, 0, 1);

  for (int i = t; i < 128 * 128; i += 256) {
    int ch = i >> 7, p = i & 127;
    as[p * ASTR + ch] = att[((size_t)b * HID + ch) * NPOS + pbase + p];
  }
  __syncthreads();

  const int cofs = (lane & 16) ? 8 : 0;
  const int arow = wave * 16 + (lane & 15);
  v16bf afrag[4];
#pragma unroll
  for (int kbk = 0; kbk < 4; ++kbk) {
    const __bf16* ap = &as[arow * ASTR + kbk * 32 + cofs];
    afrag[kbk] = cat8(*(const v8bf*)ap, *(const v8bf*)(ap + 16));
  }
  const int n_ = lane & 15, mbase = (lane >> 4) * 8;
  for (int ob = 0; ob < 16; ++ob) {
    v8f acc = {0.f, 0.f, 0.f, 0.f, 0.f, 0.f, 0.f, 0.f};
#pragma unroll
    for (int kbk = 0; kbk < 4; ++kbk) {
      const __bf16* bp = wo + ((size_t)((ob * 4 + kbk) * 32 + lane)) * 16;
      v16bf bm = cat8(*(const v8bf*)bp, *(const v8bf*)(bp + 8));
      acc = __builtin_amdgcn_wmma_f32_16x16x32_bf16(false, afrag[kbk], false, bm,
                                                    (short)0, acc, false, false);
    }
    float bias = bout[ob * 16 + n_];
    __bf16* outp = y + ((size_t)b * CDIM + ob * 16 + n_) * NPOS
                     + pbase + wave * 16 + mbase;
#pragma unroll
    for (int r = 0; r < 8; ++r) outp[r] = (__bf16)(acc[r] + bias);
  }
}

// ---------------------------------------------------------------------------
// Kernel 4: final RMSNorm(y)*g2 -> f32 d_out. Grid (32,32), 256 threads.
// ---------------------------------------------------------------------------
__global__ void la_rms2(const __bf16* __restrict__ y,
                        const float* __restrict__ g2,
                        float* __restrict__ out) {
  const int b = blockIdx.y, tile = blockIdx.x;
  const int t = threadIdx.x;
  const int pbase = tile * 128;
  __shared__ float red[256];
  __shared__ float rinv[128];
  const int px = t & 127, half = t >> 7;
  const __bf16* yb = y + (size_t)b * CDIM * NPOS + pbase + px;
  float s = 0.f;
  for (int c = half * 128; c < half * 128 + 128; ++c) {
    float v = (float)yb[(size_t)c * NPOS];
    s += v * v;
  }
  red[t] = s;
  __syncthreads();
  if (half == 0)
    rinv[px] = 16.0f / fmaxf(sqrtf(red[px] + red[px + 128]), 1e-12f);
  __syncthreads();
  float r = rinv[px];
  for (int c = half * 128; c < half * 128 + 128; ++c)
    out[((size_t)b * CDIM + c) * NPOS + pbase + px] =
        (float)yb[(size_t)c * NPOS] * r * g2[c];
}

// ---------------------------------------------------------------------------
extern "C" void kernel_launch(void* const* d_in, const int* in_sizes, int n_in,
                              void* d_out, int out_size, void* d_ws, size_t ws_size,
                              hipStream_t stream) {
  const float* x     = (const float*)d_in[0];   // (32,256,64,64)
  const float* g1    = (const float*)d_in[1];   // (256,)
  const float* memkv = (const float*)d_in[2];   // (2,4,32,4)
  const float* Wqkv  = (const float*)d_in[3];   // (384,256)
  const float* Wout  = (const float*)d_in[4];   // (256,128)
  const float* bout  = (const float*)d_in[5];   // (256,)
  const float* g2    = (const float*)d_in[6];   // (256,)
  float* out = (float*)d_out;

  char* ws = (char*)d_ws;
  __bf16* wq  = (__bf16*)ws;                               // 384*256 bf16
  __bf16* wo  = (__bf16*)(ws + 196608);                    // 256*128 bf16
  __bf16* qkv = (__bf16*)(ws + 262144);                    // 32*384*4096 bf16 (96 MB)
  __bf16* att = (__bf16*)(ws + 262144 + 100663296ull);     // 32*128*4096 bf16 (32 MB)
  __bf16* yb  = qkv;   // y (32*256*4096 bf16, 64 MB) aliases dead qkv region

  // la_qkv uses 69120 B dynamic LDS (>64K default cap; WGP has 320K).
  (void)hipFuncSetAttribute((const void*)la_qkv,
                            hipFuncAttributeMaxDynamicSharedMemorySize, 69120);

  la_prep_weights<<<32, 256, 0, stream>>>(Wqkv, Wout, wq, wo);
  la_qkv<<<dim3(32, 32), 256, 69120, stream>>>(x, g1, wq, qkv);
  la_attn<<<dim3(HEADS, NBATCH), 256, 46336, stream>>>(qkv, memkv, att);
  la_proj<<<dim3(32, 32), 256, 34816, stream>>>(att, wo, bout, yb);
  la_rms2<<<dim3(32, 32), 256, 0, stream>>>(yb, g2, out);
}
